// OutputOBIE_9053791060585
// MI455X (gfx1250) — compile-verified, matrix operand order checked by reading, not run on hardware
//
#include <hip/hip_runtime.h>
#include <hip/hip_bf16.h>

// Problem constants (from reference)
#define Bsz 64
#define Tlen 512
#define Dim 1024
#define NUM_TYPES 14
#define NUM_TAGS 4
#define NOUT 56                 // NUM_TYPES*NUM_TAGS
#define NPAD 64                 // padded N (4 tiles of 16)
#define Mrows (Bsz * Tlen)      // 32768
#define NSEQ (Bsz * NUM_TYPES)  // 896
#define EM_SIZE ((size_t)Bsz * NUM_TYPES * Tlen * NUM_TAGS)  // 1,835,008 floats

typedef __attribute__((ext_vector_type(16))) __bf16 v16bf;
typedef __attribute__((ext_vector_type(8)))  __bf16 v8bf;
typedef __attribute__((ext_vector_type(8)))  float  v8f;
typedef __attribute__((ext_vector_type(4)))  float  v4f;

// ---------------------------------------------------------------------------
// Kernel 0: split W (56x1024 f32) into bf16 hi/lo planes, zero-padded to 64 rows.
// x ~= hi + lo with |lo| <= ulp_bf16(hi)/2 -> bf16x3 GEMM recovers ~fp32 accuracy.
// ---------------------------------------------------------------------------
__global__ void prep_w_kernel(const float* __restrict__ W,
                              __bf16* __restrict__ whi,
                              __bf16* __restrict__ wlo) {
    int idx = blockIdx.x * 256 + threadIdx.x;       // 0 .. 64*1024-1
    int n = idx >> 10;
    int k = idx & 1023;
    float x = (n < NOUT) ? W[n * Dim + k] : 0.0f;
    __bf16 h = (__bf16)x;
    float  r = x - (float)h;
    whi[idx] = h;
    wlo[idx] = (__bf16)r;
}

// ---------------------------------------------------------------------------
// Kernel 1: emission = X(32768x1024) * W^T + b, written permuted (B,14,T,4).
// One wave per 16-row M tile; 4 N-tiles of 16 cols; K chunks of 32 with
// 3 x v_wmma_f32_16x16x32_bf16 per N-tile (bf16x3 decomposition).
// ---------------------------------------------------------------------------
__global__ void __launch_bounds__(128)
emission_gemm_kernel(const float* __restrict__ X,
                     const __bf16* __restrict__ Whi,
                     const __bf16* __restrict__ Wlo,
                     const float* __restrict__ bias,
                     float* __restrict__ out) {
    const int tid  = threadIdx.x;
    const int wave = tid >> 5;
    const int lane = tid & 31;
    const int half = lane >> 4;   // which 16-lane half
    const int lr   = lane & 15;   // lane within half
    const int m0   = (blockIdx.x * 4 + wave) * 16;

    const float* arow = X + (size_t)(m0 + lr) * Dim;

    v8f acc[4];
#pragma unroll
    for (int j = 0; j < 4; ++j) acc[j] = (v8f){};

    for (int kc = 0; kc < Dim / 32; ++kc) {
        // Per-lane K pattern for 16-bit 16x32 A (and 32x16 B) operands:
        //   V0-3 hold K = kb1+0..7, V4-7 hold K = kb2+0..7
        const int kb1 = kc * 32 + half * 8;
        const int kb2 = kc * 32 + 16 + half * 8;

        // ---- A: 16 contiguous-ish f32, converted to hi/lo bf16 once, reused 4x
        v4f a0 = *(const v4f*)(arow + kb1);
        v4f a1 = *(const v4f*)(arow + kb1 + 4);
        v4f a2 = *(const v4f*)(arow + kb2);
        v4f a3 = *(const v4f*)(arow + kb2 + 4);
        float af[16];
#pragma unroll
        for (int i = 0; i < 4; ++i) { af[i] = a0[i]; af[4+i] = a1[i]; af[8+i] = a2[i]; af[12+i] = a3[i]; }
        v16bf Ahi, Alo;
#pragma unroll
        for (int i = 0; i < 16; ++i) {
            __bf16 h = (__bf16)af[i];
            Ahi[i] = h;
            Alo[i] = (__bf16)(af[i] - (float)h);
        }

#pragma unroll
        for (int j = 0; j < 4; ++j) {
            const size_t nrow = (size_t)(j * 16 + lr) * Dim;
            v8bf bh0 = *(const v8bf*)(Whi + nrow + kb1);
            v8bf bh1 = *(const v8bf*)(Whi + nrow + kb2);
            v8bf bl0 = *(const v8bf*)(Wlo + nrow + kb1);
            v8bf bl1 = *(const v8bf*)(Wlo + nrow + kb2);
            v16bf Bhi = __builtin_shufflevector(bh0, bh1, 0,1,2,3,4,5,6,7,8,9,10,11,12,13,14,15);
            v16bf Blo = __builtin_shufflevector(bl0, bl1, 0,1,2,3,4,5,6,7,8,9,10,11,12,13,14,15);

            acc[j] = __builtin_amdgcn_wmma_f32_16x16x32_bf16(
                false, Ahi, false, Bhi, (short)0, acc[j], false, false);
            acc[j] = __builtin_amdgcn_wmma_f32_16x16x32_bf16(
                false, Ahi, false, Blo, (short)0, acc[j], false, false);
            acc[j] = __builtin_amdgcn_wmma_f32_16x16x32_bf16(
                false, Alo, false, Bhi, (short)0, acc[j], false, false);
        }
    }

    // Epilogue: C/D layout -> lane holds col n = j*16+lr, rows m0 + half*8 + r.
    // Scatter into permuted (B, 14, T, 4) layout with bias.
#pragma unroll
    for (int j = 0; j < 4; ++j) {
        const int n = j * 16 + lr;
        if (n < NOUT) {
            const float bj = bias[n];
            const int typ = n >> 2, tag = n & 3;
#pragma unroll
            for (int r = 0; r < 8; ++r) {
                const int m = m0 + half * 8 + r;
                const int bidx = m >> 9;        // m / Tlen
                const int t    = m & (Tlen - 1);
                out[(((size_t)bidx * NUM_TYPES + typ) * Tlen + t) * NUM_TAGS + tag] =
                    acc[j][r] + bj;
            }
        }
    }
}

// ---------------------------------------------------------------------------
// Kernel 2: batched Viterbi (pytorch-crf semantics), one lane per sequence.
// Backpointers packed 2 bits/tag -> 1 byte/step in workspace, coalesced over n.
// ---------------------------------------------------------------------------
__global__ void __launch_bounds__(128)
viterbi_kernel(const float* __restrict__ em_all,          // (NSEQ, Tlen, 4)
               const unsigned char* __restrict__ mask,    // (Bsz, Tlen) bool
               const float* __restrict__ start_t,
               const float* __restrict__ end_t,
               const float* __restrict__ trans,           // (4,4)
               unsigned char* __restrict__ hist,          // (Tlen-1, NSEQ)
               float* __restrict__ pred) {                // (NSEQ, Tlen)
    const int n = blockIdx.x * blockDim.x + threadIdx.x;
    if (n >= NSEQ) return;

    const float* em = em_all + (size_t)n * Tlen * NUM_TAGS;
    const unsigned char* mrow = mask + (size_t)(n / NUM_TYPES) * Tlen;

    float tr[16];
#pragma unroll
    for (int i = 0; i < 16; ++i) tr[i] = trans[i];

    float s[4];
#pragma unroll
    for (int c = 0; c < 4; ++c) s[c] = start_t[c] + em[c];

    int seq_len = mrow[0] ? 1 : 0;

    for (int t = 1; t < Tlen; ++t) {
        v4f e = *(const v4f*)(em + (size_t)t * 4);
        float ns[4];
        unsigned byte = 0;
#pragma unroll
        for (int c = 0; c < 4; ++c) {
            float best = s[0] + tr[c];   // prev = 0
            int   bi = 0;
#pragma unroll
            for (int p = 1; p < 4; ++p) {
                float v = s[p] + tr[p * 4 + c];
                if (v > best) { best = v; bi = p; }   // first-max -> jnp.argmax
            }
            ns[c] = best + e[c];
            byte |= (unsigned)bi << (2 * c);
        }
        hist[(size_t)(t - 1) * NSEQ + n] = (unsigned char)byte;
        const bool mk = mrow[t] != 0;
        seq_len += mk ? 1 : 0;
#pragma unroll
        for (int c = 0; c < 4; ++c) s[c] = mk ? ns[c] : s[c];
    }

    const int seq_end = seq_len - 1;

    // last_tag = argmax(score + end_transitions), first-max tie-break
    int   cur  = 0;
    float best = s[0] + end_t[0];
#pragma unroll
    for (int c = 1; c < 4; ++c) {
        float v = s[c] + end_t[c];
        if (v > best) { best = v; cur = c; }
    }

    for (int t = Tlen - 1; t >= 1; --t) {
        const bool valid = (t <= seq_end);
        pred[(size_t)n * Tlen + t] = valid ? (float)cur : 0.0f;
        const int bp   = hist[(size_t)(t - 1) * NSEQ + n];
        const int prev = (bp >> (2 * cur)) & 3;
        cur = valid ? prev : cur;
    }
    pred[(size_t)n * Tlen] = (float)cur;
}

// ---------------------------------------------------------------------------
extern "C" void kernel_launch(void* const* d_in, const int* in_sizes, int n_in,
                              void* d_out, int out_size, void* d_ws, size_t ws_size,
                              hipStream_t stream) {
    const float*         X      = (const float*)d_in[0];          // (B,T,D) f32
    const unsigned char* mask   = (const unsigned char*)d_in[1];  // (B,T) bool
    const float*         W      = (const float*)d_in[2];          // (56,1024) f32
    const float*         bias   = (const float*)d_in[3];          // (56,)
    const float*         startt = (const float*)d_in[4];          // (4,)
    const float*         endt   = (const float*)d_in[5];          // (4,)
    const float*         trans  = (const float*)d_in[6];          // (4,4)

    float* out = (float*)d_out;           // [emission (B,14,T,4) | pred (B,14,T)]

    // Workspace layout: Whi (64*1024 bf16 = 128KB) | Wlo (128KB) | hist (~448KB)
    __bf16* whi = (__bf16*)d_ws;
    __bf16* wlo = whi + (size_t)NPAD * Dim;
    unsigned char* hist = (unsigned char*)d_ws + 2u * NPAD * Dim * sizeof(__bf16);

    prep_w_kernel<<<(NPAD * Dim) / 256, 256, 0, stream>>>(W, whi, wlo);

    // 32768 rows / (16 rows/wave * 4 waves/block) = 512 blocks
    emission_gemm_kernel<<<Mrows / 64, 128, 0, stream>>>(X, whi, wlo, bias, out);

    viterbi_kernel<<<(NSEQ + 127) / 128, 128, 0, stream>>>(
        out, mask, startt, endt, trans, hist, out + EM_SIZE);
}
